// SelectK_83013127897583
// MI455X (gfx1250) — matrix operand order repaired and checked by reading, not compiled, not run on hardware
//
#include <hip/hip_runtime.h>
#include <hip/hip_bf16.h>

// ---------------------------------------------------------------------------
// MI455X (gfx1250) strategy:
//  * Skip the dense 34-GFLOP hidden@W2s GEMM: predictions_senses only needs
//    logits_senses at <=K*G (128) candidate columns per row. Compute those
//    with v_wmma_f32_16x16x32_bf16 using a hi/lo bf16 split (3 WMMAs/step)
//    for ~fp32 accuracy.
//  * Pre-transpose W2s [H,S] -> W2sT [S,H] in workspace (LDS-tiled) so the
//    candidate columns become contiguous 2KB rows: avoids ~9 GB of wasted
//    cache-line traffic from column-strided 4B gathers. Falls back to the
//    strided path if ws_size is too small (deterministic per ws_size).
//  * log_softmax over [N,V]: per-row block does K argmax passes + sum-exp +
//    fused output write while the 200KB row is L2-hot (no second kernel
//    re-reading 224MB from HBM).
//  * Total HBM traffic ~0.7 GB -> ~30us at 23.3 TB/s; matrix work negligible.
// ---------------------------------------------------------------------------

typedef __attribute__((ext_vector_type(16))) __bf16 v16bf;
typedef __attribute__((ext_vector_type(8)))  float  v8f;

#define KMAX_C 32
#define CMAX   1024   // max K*G candidates (K<=32, G==32)
#define HMAX   1024

__device__ __forceinline__ float neg_inf() { return -__builtin_inff(); }

// ---------------------------------------------------------------------------
// Kernel A: per-row top-K + log_softmax stats + fused globals output write.
// One block per row; the row stays L2-resident across all passes.
// ---------------------------------------------------------------------------
__global__ void topk_lsm_kernel(const float* __restrict__ lg,
                                const int* __restrict__ Kp,
                                int V, int KMAX,
                                int* __restrict__ topk,
                                float* __restrict__ outG) {
  const int row = blockIdx.x;
  const int tid = threadIdx.x;
  const int NT  = blockDim.x;   // 256
  __shared__ float sval[256];
  __shared__ int   sidx[256];
  __shared__ int   chosen[KMAX_C];
  __shared__ float smax, slse;

  const float* x = lg + (size_t)row * V;
  int K = *Kp; if (K < 1) K = 1; if (K > KMAX) K = KMAX;

  // --- K iterative argmax passes (ties -> lowest index, like lax.top_k)
  for (int k = 0; k < K; ++k) {
    float best = neg_inf(); int bidx = 0x7fffffff;
    for (int j = tid; j < V; j += NT) {
      bool skip = false;
      for (int q = 0; q < k; ++q) skip |= (chosen[q] == j);
      if (skip) continue;
      float v = x[j];
      if (v > best || (v == best && j < bidx)) { best = v; bidx = j; }
    }
    sval[tid] = best; sidx[tid] = bidx;
    __syncthreads();
    for (int s = NT >> 1; s > 0; s >>= 1) {
      if (tid < s) {
        float v2 = sval[tid + s]; int i2 = sidx[tid + s];
        if (v2 > sval[tid] || (v2 == sval[tid] && i2 < sidx[tid])) {
          sval[tid] = v2; sidx[tid] = i2;
        }
      }
      __syncthreads();
    }
    if (tid == 0) {
      chosen[k] = sidx[0];
      topk[(size_t)row * KMAX + k] = sidx[0];
      if (k == 0) smax = sval[0];
    }
    __syncthreads();
  }

  // --- log-sum-exp pass
  const float m = smax;
  float acc = 0.f;
  for (int j = tid; j < V; j += NT) acc += expf(x[j] - m);
  sval[tid] = acc;
  __syncthreads();
  for (int s = NT >> 1; s > 0; s >>= 1) {
    if (tid < s) sval[tid] += sval[tid + s];
    __syncthreads();
  }
  if (tid == 0) slse = logf(sval[0]);
  __syncthreads();

  // --- fused coalesced output write while the row is cache-hot
  const float sub = m + slse;
  float* o = outG + (size_t)row * V;
  for (int j = tid; j < V; j += NT) o[j] = x[j] - sub;
}

// ---------------------------------------------------------------------------
// Kernel T: LDS-tiled transpose  W2s [H,S] -> W2sT [S,H]
// ---------------------------------------------------------------------------
__global__ void transpose_kernel(const float* __restrict__ W, float* __restrict__ WT,
                                 int H, int S) {
  __shared__ float tile[32][33];
  const int s0 = blockIdx.x * 32;
  const int h0 = blockIdx.y * 32;
  const int tx = threadIdx.x & 31;
  const int ty = threadIdx.x >> 5;          // 256 threads = 32 x 8
  for (int r = ty; r < 32; r += 8) {
    const int h = h0 + r, s = s0 + tx;
    tile[r][tx] = (h < H && s < S) ? W[(size_t)h * S + s] : 0.f;
  }
  __syncthreads();
  for (int r = ty; r < 32; r += 8) {
    const int s = s0 + r, h = h0 + tx;
    if (s < S && h < H) WT[(size_t)s * H + h] = tile[tx][r];
  }
}

// ---------------------------------------------------------------------------
// Kernel C: candidate logits via hi/lo-split bf16 WMMA + softmax + scatter.
// One block (256 threads = 8 waves) per row; each wave owns a 16-cand tile.
// ---------------------------------------------------------------------------
__global__ void senses_kernel(const float* __restrict__ hidden,
                              const float* __restrict__ W2s,   // [H,S] row-major
                              const float* __restrict__ WT,    // [S,H] or null
                              const float* __restrict__ b2s,
                              const int* __restrict__ grapharea,
                              const int* __restrict__ lemma_map,
                              const int* __restrict__ Kp,
                              const int* __restrict__ topk,
                              float* __restrict__ out2,
                              int S, int H, int G, int KMAX) {
  const int row  = blockIdx.x;
  const int tid  = threadIdx.x;
  const int NT   = blockDim.x;     // 256
  const int wave = tid >> 5;
  const int lane = tid & 31;

  __shared__ __bf16 hhi[HMAX];
  __shared__ __bf16 hlo[HMAX];
  __shared__ int    candIdx[CMAX];
  __shared__ float  candVal[CMAX];
  __shared__ float  red[256];
  __shared__ int    redi[256];
  __shared__ float  s_max, s_sum, s_delta;
  __shared__ int    s_amax;

  int K = *Kp; if (K < 1) K = 1; if (K > KMAX) K = KMAX;
  int C = K * G; if (C > CMAX) C = CMAX;
  const int nTiles = (C + 15) >> 4;
  const int Cpad   = nTiles << 4;
  const int Hm     = H & ~31;      // WMMA-covered K extent

  // hidden row -> LDS as bf16 hi/lo split
  for (int j = tid; j < H; j += NT) {
    const float x = hidden[(size_t)row * H + j];
    const __bf16 h = (__bf16)x;
    hhi[j] = h;
    hlo[j] = (__bf16)(x - (float)h);
  }

  // candidate list: neigh = grapharea[lemma_map[topk_k]][g] - 1
  for (int j = tid; j < Cpad; j += NT) {
    int idx = -1;
    if (j < C) {
      const int k  = j / G, g = j - k * G;
      const int gl = topk[(size_t)row * KMAX + k];
      const int lm = lemma_map[gl];
      const int v  = grapharea[(size_t)lm * G + g] - 1;
      if (v >= 0 && v < S) idx = v;
    }
    candIdx[j] = idx;
    candVal[j] = neg_inf();
  }
  __syncthreads();

  // --- candidate logits: D = A(bcast hidden row 16x32) x B(cand cols 32x16)
  for (int tile = wave; tile < nTiles; tile += (NT >> 5)) {
    v8f acc = {};
    const int n       = lane & 15;
    const int col     = candIdx[(tile << 4) + n];
    const int safecol = (col < 0) ? 0 : col;
    const int hi      = (lane >= 16);
    const int aoff0   = hi ? 16 : 0;    // A elems 0..7  : K = {0..7 | 16..23}
    const int aoff1   = hi ? 24 : 8;    // A elems 8..15 : K = {8..15 | 24..31}
    const int koff    = hi ? 16 : 0;    // B lanes 0-15: K=0..15, 16-31: K=16..31

    if (WT != nullptr) {
      // contiguous column read from transposed weights (4x b128 per step)
      const float* wrow = WT + (size_t)safecol * H;
#pragma unroll 2
      for (int k0 = 0; k0 < Hm; k0 += 32) {
        float wf[16];
        const float4* w4 = (const float4*)(wrow + k0 + koff);
#pragma unroll
        for (int q = 0; q < 4; ++q) {
          const float4 w = w4[q];
          wf[4*q+0] = w.x; wf[4*q+1] = w.y; wf[4*q+2] = w.z; wf[4*q+3] = w.w;
        }
        v16bf ah, al, bh, bl;
#pragma unroll
        for (int i = 0; i < 8; ++i) {
          ah[i] = hhi[k0 + aoff0 + i];  ah[i+8] = hhi[k0 + aoff1 + i];
          al[i] = hlo[k0 + aoff0 + i];  al[i+8] = hlo[k0 + aoff1 + i];
        }
#pragma unroll
        for (int i = 0; i < 16; ++i) {
          const __bf16 h = (__bf16)wf[i];
          bh[i] = h;
          bl[i] = (__bf16)(wf[i] - (float)h);
        }
        acc = __builtin_amdgcn_wmma_f32_16x16x32_bf16(false, ah, false, bh, (short)0, acc, false, false);
        acc = __builtin_amdgcn_wmma_f32_16x16x32_bf16(false, ah, false, bl, (short)0, acc, false, false);
        acc = __builtin_amdgcn_wmma_f32_16x16x32_bf16(false, al, false, bh, (short)0, acc, false, false);
      }
    } else {
      // fallback: column-strided gather from row-major W2s
      for (int k0 = 0; k0 < Hm; k0 += 32) {
        float wf[16];
#pragma unroll
        for (int i = 0; i < 16; ++i)
          wf[i] = W2s[(size_t)(k0 + koff + i) * S + safecol];
        v16bf ah, al, bh, bl;
#pragma unroll
        for (int i = 0; i < 8; ++i) {
          ah[i] = hhi[k0 + aoff0 + i];  ah[i+8] = hhi[k0 + aoff1 + i];
          al[i] = hlo[k0 + aoff0 + i];  al[i+8] = hlo[k0 + aoff1 + i];
        }
#pragma unroll
        for (int i = 0; i < 16; ++i) {
          const __bf16 h = (__bf16)wf[i];
          bh[i] = h;
          bl[i] = (__bf16)(wf[i] - (float)h);
        }
        acc = __builtin_amdgcn_wmma_f32_16x16x32_bf16(false, ah, false, bh, (short)0, acc, false, false);
        acc = __builtin_amdgcn_wmma_f32_16x16x32_bf16(false, ah, false, bl, (short)0, acc, false, false);
        acc = __builtin_amdgcn_wmma_f32_16x16x32_bf16(false, al, false, bh, (short)0, acc, false, false);
      }
    }

    if (lane < 16 && col >= 0) {
      float logit = acc[0];                                   // M=0 row, N=lane
      for (int k = Hm; k < H; ++k)                            // fp32 scalar tail
        logit += hidden[(size_t)row * H + k] * W2s[(size_t)k * S + safecol];
      candVal[(tile << 4) + n] = logit + b2s[safecol];
    }
  }
  __syncthreads();

  // --- softmax over candidates: max/argmax (first-index ties)
  {
    float bmax = neg_inf(); int bidx = 0x7fffffff;
    for (int j = tid; j < C; j += NT) {
      float v = candVal[j];
      if (v > bmax || (v == bmax && j < bidx)) { bmax = v; bidx = j; }
    }
    red[tid] = bmax; redi[tid] = bidx;
    __syncthreads();
    for (int s = NT >> 1; s > 0; s >>= 1) {
      if (tid < s) {
        float v2 = red[tid + s]; int i2 = redi[tid + s];
        if (v2 > red[tid] || (v2 == red[tid] && i2 < redi[tid])) {
          red[tid] = v2; redi[tid] = i2;
        }
      }
      __syncthreads();
    }
    if (tid == 0) { s_max = red[0]; s_amax = redi[0]; }
    __syncthreads();
  }

  // --- sum(exp) and n_valid
  {
    float a = 0.f; int cnt = 0;
    for (int j = tid; j < C; j += NT) {
      if (candIdx[j] >= 0) { a += expf(candVal[j] - s_max); cnt++; }
    }
    red[tid] = a; redi[tid] = cnt;
    __syncthreads();
    for (int s = NT >> 1; s > 0; s >>= 1) {
      if (tid < s) { red[tid] += red[tid + s]; redi[tid] += redi[tid + s]; }
      __syncthreads();
    }
    if (tid == 0) {
      s_sum   = red[0];
      s_delta = 1e-8f * (float)(S - redi[0]);
    }
    __syncthreads();
  }

  // --- fill log(EPS) (vectorized), then scatter log(p)
  const float LOG_EPS = -18.420680743952367f;   // log(1e-8)
  float* orow = out2 + (size_t)row * S;
  if (((S & 3) == 0) && ((((size_t)orow) & 15) == 0)) {
    float4* o4 = (float4*)orow;
    const float4 fe = make_float4(LOG_EPS, LOG_EPS, LOG_EPS, LOG_EPS);
    const int S4 = S >> 2;
    for (int j = tid; j < S4; j += NT) o4[j] = fe;
  } else {
    for (int j = tid; j < S; j += NT) orow[j] = LOG_EPS;
  }
  __syncthreads();
  for (int j = tid; j < C; j += NT) {
    const int col = candIdx[j];
    if (col >= 0) {
      float p = expf(candVal[j] - s_max) / s_sum;
      if (j == s_amax) p -= s_delta;
      orow[col] = logf(p);
    }
  }
}

// ---------------------------------------------------------------------------
extern "C" void kernel_launch(void* const* d_in, const int* in_sizes, int n_in,
                              void* d_out, int out_size, void* d_ws, size_t ws_size,
                              hipStream_t stream) {
  const float* lg        = (const float*)d_in[0];   // [N,V]
  const float* hidden    = (const float*)d_in[1];   // [N,H]
  const float* W2s       = (const float*)d_in[2];   // [H,S]
  const float* b2s       = (const float*)d_in[3];   // [S]
  const int*   grapharea = (const int*)d_in[4];     // [V,G]
  const int*   lemma_map = (const int*)d_in[5];     // [V]
  const int*   Kp        = (const int*)d_in[6];     // scalar

  const int S = in_sizes[3];
  const int V = in_sizes[5];
  const int H = in_sizes[2] / S;
  const int N = in_sizes[1] / H;
  const int G = in_sizes[4] / V;
  const int KMAX = KMAX_C;

  // workspace layout: topk[N*KMAX] | (aligned) W2sT[S*H]
  int* topk = (int*)d_ws;
  size_t base = (size_t)N * KMAX * sizeof(int);
  base = (base + 255) & ~(size_t)255;
  const size_t wtBytes = (size_t)H * S * sizeof(float);
  float* WT = nullptr;
  if ((H & 3) == 0 && ws_size >= base + wtBytes)
    WT = (float*)((char*)d_ws + base);

  float* outG = (float*)d_out;                     // [N,V]
  float* outS = outG + (size_t)N * V;              // [N,S]

  topk_lsm_kernel<<<N, 256, 0, stream>>>(lg, Kp, V, KMAX, topk, outG);

  if (WT) {
    dim3 gT((S + 31) / 32, (H + 31) / 32);
    transpose_kernel<<<gT, 256, 0, stream>>>(W2s, WT, H, S);
  }

  senses_kernel<<<N, 256, 0, stream>>>(hidden, W2s, WT, b2s, grapharea, lemma_map,
                                       Kp, topk, outS, S, H, G, KMAX);
}
